// ReMalAttNet_75728863363480
// MI455X (gfx1250) — matrix operand order repaired and checked by reading, not compiled, not run on hardware
//
#include <hip/hip_runtime.h>
#include <math.h>

typedef __attribute__((ext_vector_type(16))) _Float16 v16h;
typedef __attribute__((ext_vector_type(8)))  float    v8f;
typedef __attribute__((ext_vector_type(4)))  unsigned int u32x4;
typedef __attribute__((ext_vector_type(8)))  int      i32x8;
typedef __attribute__((ext_vector_type(4)))  int      i32x4;

#define BN_INV 0.9999950000374996f  /* 1/sqrt(1+1e-5) */

__device__ __forceinline__ v8f wmma32(v16h a, v16h b, v8f c) {
  // D = A(16x32 f16) * B(32x16 f16) + C(16x16 f32)
  return __builtin_amdgcn_wmma_f32_16x16x32_f16(false, a, false, b, (short)0, c, false, false);
}
// A-fragment element e (0..15) of lane-half hi -> K offset within 32-chunk
__device__ __forceinline__ int akof(int e, int hi) { return (hi << 4) + e; }

__device__ __forceinline__ float sigmoidf_(float x) { return 1.f / (1.f + expf(-x)); }

// ---------------- Block 1: direct conv(1->32,s2) + BN + ReLU + pool2 ----------------
__global__ void k_conv1_pool(const float* __restrict__ support, const float* __restrict__ query,
                             const float* __restrict__ w, const float* __restrict__ bias,
                             const float* __restrict__ g, const float* __restrict__ be,
                             float* __restrict__ out) {
  int idx = blockIdx.x * blockDim.x + threadIdx.x;
  const int total = 40 * 32 * 56 * 56;
  if (idx >= total) return;
  int ox = idx % 56, oy = (idx / 56) % 56, co = (idx / (56 * 56)) % 32, b = idx / (56 * 56 * 32);
  const float* in = (b < 25) ? (support + (size_t)b * 224 * 224)
                             : (query + (size_t)(b - 25) * 224 * 224);
  float scale = g[co] * BN_INV, beta = be[co], bb = bias[co];
  float mx = -__builtin_inff();
  for (int dy = 0; dy < 2; ++dy)
    for (int dx = 0; dx < 2; ++dx) {
      int cy = 2 * oy + dy, cx = 2 * ox + dx;  // conv-out coords (112x112)
      float acc = bb;
      for (int r = 0; r < 3; ++r)
        for (int s = 0; s < 3; ++s) {
          int iy = 2 * cy + r - 1, ix = 2 * cx + s - 1;
          if (iy >= 0 && iy < 224 && ix >= 0 && ix < 224)
            acc += w[co * 9 + r * 3 + s] * in[iy * 224 + ix];
        }
      float v = fmaxf(acc * scale + beta, 0.f);
      mx = fmaxf(mx, v);
    }
  out[idx] = mx;
}

// ---------------- Implicit-GEMM conv (stride1,pad1) + BN + ReLU, WMMA ----------------
// one wave per 16(cout) x 32(pixel) tile: A(weights) fragment reused for 2 pixel tiles
__global__ void k_conv_wmma(const float* __restrict__ in, const float* __restrict__ w,
                            const float* __restrict__ bias, const float* __restrict__ g,
                            const float* __restrict__ be, float* __restrict__ out,
                            int Cin, int Cout, int H, int W) {
  const int HW = H * W;
  const int K = Cin * 9;
  int p0 = blockIdx.x * 32, m0 = blockIdx.y * 16, b = blockIdx.z;
  int lane = threadIdx.x, li = lane & 15, hi = lane >> 4;
  int pA = p0 + li, pB = p0 + 16 + li;
  bool vA = pA < HW, vB = pB < HW;
  int yA = vA ? pA / W : 0, xA = vA ? pA - (pA / W) * W : 0;
  int yB = vB ? pB / W : 0, xB = vB ? pB - (pB / W) * W : 0;
  const float* inB = in + (size_t)b * Cin * HW;
  v8f acc0 = {}, acc1 = {};
  for (int kc = 0; kc < K; kc += 32) {
    v16h a, b0, b1;
#pragma unroll
    for (int e = 0; e < 16; ++e) {
      int ka = kc + akof(e, hi);
      a[e] = (_Float16)w[(size_t)(m0 + li) * K + ka];  // weights already [Cout][Cin*9]
    }
#pragma unroll
    for (int e = 0; e < 16; ++e) {
      int kb = kc + hi * 16 + e;
      int ci = kb / 9, rs = kb - ci * 9;
      int r = rs / 3, s = rs - (rs / 3) * 3;
      size_t base = (size_t)ci * HW;
      int iyA = yA + r - 1, ixA = xA + s - 1;
      int iyB = yB + r - 1, ixB = xB + s - 1;
      float fa = 0.f, fb = 0.f;
      if (vA && iyA >= 0 && iyA < H && ixA >= 0 && ixA < W) fa = inB[base + iyA * W + ixA];
      if (vB && iyB >= 0 && iyB < H && ixB >= 0 && ixB < W) fb = inB[base + iyB * W + ixB];
      b0[e] = (_Float16)fa;
      b1[e] = (_Float16)fb;
    }
    acc0 = wmma32(a, b0, acc0);
    acc1 = wmma32(a, b1, acc1);
  }
#pragma unroll
  for (int r8 = 0; r8 < 8; ++r8) {
    int m = m0 + hi * 8 + r8;
    float sc = g[m] * BN_INV, bt = be[m], bb = bias[m];
    if (vA) out[((size_t)b * Cout + m) * HW + pA] = fmaxf((acc0[r8] + bb) * sc + bt, 0.f);
    if (vB) out[((size_t)b * Cout + m) * HW + pB] = fmaxf((acc1[r8] + bb) * sc + bt, 0.f);
  }
}

// ---------------- 2x2 max pool ----------------
__global__ void k_pool2(const float* __restrict__ in, float* __restrict__ out,
                        int C, int H, int W, int B) {
  int OH = H / 2, OW = W / 2;
  int idx = blockIdx.x * blockDim.x + threadIdx.x;
  int total = B * C * OH * OW;
  if (idx >= total) return;
  int ox = idx % OW, oy = (idx / OW) % OH, c = (idx / (OW * OH)) % C, b = idx / (OW * OH * C);
  const float* p = in + ((size_t)b * C + c) * H * W + (2 * oy) * W + 2 * ox;
  out[idx] = fmaxf(fmaxf(p[0], p[1]), fmaxf(p[W], p[W + 1]));
}

// ---------------- Attention projection: OUT[m][h] = feat(m) . att_w[h] + att_b ----------------
__global__ void k_att_proj(const float* __restrict__ pool4, const float* __restrict__ aw,
                           const float* __restrict__ ab, float* __restrict__ out,
                           int b0, int Mrows) {
  int n0 = blockIdx.x * 16, m0 = blockIdx.y * 16;
  int lane = threadIdx.x, li = lane & 15, hi = lane >> 4;
  int ma = m0 + li;
  bool mv = ma < Mrows;
  const float* Arow = pool4 + ((size_t)(b0 + (mv ? ma : 0) / 49) * 256 * 49 + ((mv ? ma : 0) % 49));
  v8f acc = {};
  for (int kc = 0; kc < 256; kc += 32) {
    v16h a, bf;
#pragma unroll
    for (int e = 0; e < 16; ++e) {
      int ka = kc + akof(e, hi);
      a[e] = (_Float16)(mv ? Arow[(size_t)ka * 49] : 0.f);
    }
#pragma unroll
    for (int e = 0; e < 16; ++e) {
      int kb = kc + hi * 16 + e;
      bf[e] = (_Float16)aw[(size_t)(n0 + li) * 256 + kb];
    }
    acc = wmma32(a, bf, acc);
  }
#pragma unroll
  for (int r = 0; r < 8; ++r) {
    int m = m0 + hi * 8 + r;
    if (m < Mrows) out[(size_t)m * 64 + (n0 + li)] = acc[r] + ab[n0 + li];
  }
}

// ---------------- scores[q,s,t,u] = sum_h tanh(Wq*Wh) ----------------
__global__ void k_scores(const float* __restrict__ WQ, const float* __restrict__ WH,
                         float* __restrict__ SC) {
  int idx = blockIdx.x * blockDim.x + threadIdx.x;
  const int total = 15 * 25 * 49 * 49;
  if (idx >= total) return;
  int u = idx % 49, t = (idx / 49) % 49, s = (idx / 2401) % 25, q = idx / (2401 * 25);
  const float* wq = WQ + (size_t)(q * 49 + t) * 64;
  const float* wh = WH + (size_t)(s * 49 + u) * 64;
  float acc = 0.f;
#pragma unroll 4
  for (int h = 0; h < 64; ++h) acc += tanhf(wq[h] * wh[h]);
  SC[idx] = acc;
}

// ---------------- softmax over rows of length 49 (one wave per row) ----------------
__global__ void k_softmax49(float* __restrict__ SC) {
  int row = blockIdx.x;
  float* p = SC + (size_t)row * 49;
  int l = threadIdx.x;
  float v0 = p[l];
  float v1 = (l + 32 < 49) ? p[l + 32] : -__builtin_inff();
  float m = fmaxf(v0, v1);
  for (int o = 16; o > 0; o >>= 1) m = fmaxf(m, __shfl_xor(m, o, 32));
  float e0 = expf(v0 - m), e1 = (l + 32 < 49) ? expf(v1 - m) : 0.f;
  float sum = e0 + e1;
  for (int o = 16; o > 0; o >>= 1) sum += __shfl_xor(sum, o, 32);
  float inv = 1.f / sum;
  p[l] = e0 * inv;
  if (l + 32 < 49) p[l + 32] = e1 * inv;
}

// ---------------- attended[qs] = att(49x49) @ hf_s(49x256), WMMA with K padded to 64 ----------------
__global__ void k_attended(const float* __restrict__ SC, const float* __restrict__ pool4,
                           float* __restrict__ out) {
  int n0 = blockIdx.x * 16, m0 = blockIdx.y * 16, qs = blockIdx.z;
  int s = qs % 25;
  int lane = threadIdx.x, li = lane & 15, hi = lane >> 4;
  const float* att = SC + (size_t)qs * 49 * 49;
  const float* hfS = pool4 + (size_t)s * 256 * 49;  // [f][u]
  v8f acc = {};
  int ma = m0 + li;
  for (int kc = 0; kc < 64; kc += 32) {
    v16h a, bf;
#pragma unroll
    for (int e = 0; e < 16; ++e) {
      int ka = kc + akof(e, hi);
      a[e] = (_Float16)((ma < 49 && ka < 49) ? att[ma * 49 + ka] : 0.f);
    }
#pragma unroll
    for (int e = 0; e < 16; ++e) {
      int kb = kc + hi * 16 + e;
      bf[e] = (_Float16)((kb < 49) ? hfS[(size_t)(n0 + li) * 49 + kb] : 0.f);
    }
    acc = wmma32(a, bf, acc);
  }
#pragma unroll
  for (int r = 0; r < 8; ++r) {
    int m = m0 + hi * 8 + r;
    if (m < 49) out[((size_t)qs * 49 + m) * 256 + n0 + li] = acc[r];
  }
}

// ---------------- X = concat(attended, qrep) -> (375*49, 512) ----------------
__global__ void k_build_x(const float* __restrict__ attended, const float* __restrict__ pool4,
                          float* __restrict__ X) {
  size_t idx = (size_t)blockIdx.x * blockDim.x + threadIdx.x;
  const size_t total = (size_t)375 * 49 * 512;
  if (idx >= total) return;
  int j = (int)(idx & 511);
  size_t bt = idx >> 9;
  int t = (int)(bt % 49);
  int b = (int)(bt / 49);
  int q = b / 25;
  float v;
  if (j < 256) v = attended[bt * 256 + j];
  else         v = pool4[((size_t)(25 + q) * 256 + (j - 256)) * 49 + t];
  X[idx] = v;
}

// ---------------- GI0 = X @ wi0^T + bi0 (18375x384, K=512) ----------------
// Block = 8 waves. TDM loads the 16x512 A-panel into LDS once (zero-fill OOB rows),
// each wave computes 3 N-tiles reusing one A fragment per K-chunk.
__global__ __launch_bounds__(256) void k_gi0(const float* __restrict__ X,
                                             const float* __restrict__ wi0,
                                             const float* __restrict__ bi0,
                                             float* __restrict__ GI0) {
  __shared__ float Atile[16 * 512];
  int m0 = blockIdx.x * 16;
  int tid = threadIdx.x;
  int wave = tid >> 5, lane = tid & 31, li = lane & 15, hi = lane >> 4;
  if (wave == 0) {
    unsigned long long ga = (unsigned long long)(uintptr_t)(X + (size_t)m0 * 512);
    unsigned int lds = (unsigned int)(uintptr_t)(&Atile[0]);
    unsigned int rows = 18375u - (unsigned int)m0;
    if (rows > 16u) rows = 16u;
    // D# group0: count=1 | lds_addr | global_addr[56:0] | type=2
    u32x4 g0 = { 1u, lds, (unsigned int)ga,
                 (unsigned int)((ga >> 32) & 0x01FFFFFFu) | (2u << 30) };
    // D# group1: data_size=4B; tensor 512 x rows (stride 512); tile 512 x 16
    i32x8 g1 = { (int)(2u << 16),     // [17:16] data_size=2 (4 bytes)
                 (int)(512u << 16),   // [63:48] tensor_dim0 lo16 = 512
                 (int)(rows << 16),   // [79:64] dim0 hi16 = 0 | [95:80] tensor_dim1 lo16
                 (int)(512u << 16),   // [111:96] dim1 hi16 = 0 | [127:112] tile_dim0 = 512
                 16,                  // [143:128] tile_dim1=16 | tile_dim2=0
                 512,                 // tensor_dim0_stride[31:0] = 512 elements
                 0, 0 };
    i32x4 gz = { 0, 0, 0, 0 };
    i32x8 gz8 = { 0, 0, 0, 0, 0, 0, 0, 0 };
    __builtin_amdgcn_tensor_load_to_lds(g0, g1, gz, gz, gz8, 0);
    __builtin_amdgcn_s_wait_tensorcnt(0);
  }
  __syncthreads();
  int nb = wave * 48;
  v8f acc0 = {}, acc1 = {}, acc2 = {};
  for (int kc = 0; kc < 512; kc += 32) {
    v16h a, b0, b1, b2;
#pragma unroll
    for (int e = 0; e < 16; ++e) { int ka = kc + akof(e, hi); a[e] = (_Float16)Atile[li * 512 + ka]; }
#pragma unroll
    for (int e = 0; e < 16; ++e) {
      int kb = kc + hi * 16 + e;
      b0[e] = (_Float16)wi0[(size_t)(nb + li) * 512 + kb];
      b1[e] = (_Float16)wi0[(size_t)(nb + 16 + li) * 512 + kb];
      b2[e] = (_Float16)wi0[(size_t)(nb + 32 + li) * 512 + kb];
    }
    acc0 = wmma32(a, b0, acc0);
    acc1 = wmma32(a, b1, acc1);
    acc2 = wmma32(a, b2, acc2);
  }
#pragma unroll
  for (int r = 0; r < 8; ++r) {
    int m = m0 + hi * 8 + r;
    if (m < 18375) {
      GI0[(size_t)m * 384 + nb + li]      = acc0[r] + bi0[nb + li];
      GI0[(size_t)m * 384 + nb + 16 + li] = acc1[r] + bi0[nb + 16 + li];
      GI0[(size_t)m * 384 + nb + 32 + li] = acc2[r] + bi0[nb + 32 + li];
    }
  }
}

// ---------------- Fused 2-layer GRU recurrence: 16 sequences per workgroup ----------------
__global__ __launch_bounds__(256) void k_gru(const float* __restrict__ GI0,
    const float* __restrict__ wh0, const float* __restrict__ bh0,
    const float* __restrict__ wi1, const float* __restrict__ bi1,
    const float* __restrict__ wh1, const float* __restrict__ bh1,
    float* __restrict__ HLAST) {
  __shared__ float h0[16 * 128];
  __shared__ float h1[16 * 128];
  __shared__ float gA[16 * 384];
  __shared__ float gB[16 * 384];
  int tid = threadIdx.x;
  int wave = tid >> 5, lane = tid & 31, li = lane & 15, hi = lane >> 4;
  int bbase = blockIdx.x * 16;
  for (int i = tid; i < 16 * 128; i += 256) { h0[i] = 0.f; h1[i] = 0.f; }
  __syncthreads();
  for (int t = 0; t < 49; ++t) {
    // GH0 = h0 @ wh0^T + bh0 -> gA: wave handles 3 n-tiles sharing one A fragment
    {
      int n0 = wave * 48;
      v8f c0 = {}, c1 = {}, c2 = {};
      for (int kc = 0; kc < 128; kc += 32) {
        v16h af, b0, b1, b2;
#pragma unroll
        for (int e = 0; e < 16; ++e) { int ka = kc + akof(e, hi); af[e] = (_Float16)h0[li * 128 + ka]; }
#pragma unroll
        for (int e = 0; e < 16; ++e) {
          int kb = kc + hi * 16 + e;
          b0[e] = (_Float16)wh0[(size_t)(n0 + li) * 128 + kb];
          b1[e] = (_Float16)wh0[(size_t)(n0 + 16 + li) * 128 + kb];
          b2[e] = (_Float16)wh0[(size_t)(n0 + 32 + li) * 128 + kb];
        }
        c0 = wmma32(af, b0, c0); c1 = wmma32(af, b1, c1); c2 = wmma32(af, b2, c2);
      }
#pragma unroll
      for (int r = 0; r < 8; ++r) {
        int m = hi * 8 + r;
        gA[m * 384 + n0 + li]      = c0[r] + bh0[n0 + li];
        gA[m * 384 + n0 + 16 + li] = c1[r] + bh0[n0 + 16 + li];
        gA[m * 384 + n0 + 32 + li] = c2[r] + bh0[n0 + 32 + li];
      }
    }
    __syncthreads();
    // layer-0 gates
    for (int i = tid; i < 16 * 128; i += 256) {
      int m = i >> 7, jj = i & 127;
      int b = bbase + m;
      float ir = 0.f, iz = 0.f, inn = 0.f;
      if (b < 375) {
        const float* gi = GI0 + ((size_t)b * 49 + t) * 384;
        ir = gi[jj]; iz = gi[jj + 128]; inn = gi[jj + 256];
      }
      float r = sigmoidf_(ir + gA[m * 384 + jj]);
      float z = sigmoidf_(iz + gA[m * 384 + jj + 128]);
      float nn = tanhf(inn + r * gA[m * 384 + jj + 256]);
      h0[i] = (1.f - z) * nn + z * h0[i];
    }
    __syncthreads();
    // GI1 = h0 @ wi1^T + bi1 -> gA (waves 0-3); GH1 = h1 @ wh1^T + bh1 -> gB (waves 4-7)
    {
      bool first = wave < 4;
      const float* Wm = first ? wi1 : wh1;
      const float* Bs = first ? bi1 : bh1;
      const float* Hs = first ? h0 : h1;
      float*       Gd = first ? gA : gB;
      int wq = first ? wave : wave - 4;
      for (int half = 0; half < 2; ++half) {
        int n0 = wq * 96 + half * 48;
        v8f c0 = {}, c1 = {}, c2 = {};
        for (int kc = 0; kc < 128; kc += 32) {
          v16h af, b0, b1, b2;
#pragma unroll
          for (int e = 0; e < 16; ++e) { int ka = kc + akof(e, hi); af[e] = (_Float16)Hs[li * 128 + ka]; }
#pragma unroll
          for (int e = 0; e < 16; ++e) {
            int kb = kc + hi * 16 + e;
            b0[e] = (_Float16)Wm[(size_t)(n0 + li) * 128 + kb];
            b1[e] = (_Float16)Wm[(size_t)(n0 + 16 + li) * 128 + kb];
            b2[e] = (_Float16)Wm[(size_t)(n0 + 32 + li) * 128 + kb];
          }
          c0 = wmma32(af, b0, c0); c1 = wmma32(af, b1, c1); c2 = wmma32(af, b2, c2);
        }
#pragma unroll
        for (int r = 0; r < 8; ++r) {
          int m = hi * 8 + r;
          Gd[m * 384 + n0 + li]      = c0[r] + Bs[n0 + li];
          Gd[m * 384 + n0 + 16 + li] = c1[r] + Bs[n0 + 16 + li];
          Gd[m * 384 + n0 + 32 + li] = c2[r] + Bs[n0 + 32 + li];
        }
      }
    }
    __syncthreads();
    // layer-1 gates
    for (int i = tid; i < 16 * 128; i += 256) {
      int m = i >> 7, jj = i & 127;
      float r = sigmoidf_(gA[m * 384 + jj] + gB[m * 384 + jj]);
      float z = sigmoidf_(gA[m * 384 + jj + 128] + gB[m * 384 + jj + 128]);
      float nn = tanhf(gA[m * 384 + jj + 256] + r * gB[m * 384 + jj + 256]);
      h1[i] = (1.f - z) * nn + z * h1[i];
    }
    __syncthreads();
  }
  for (int i = tid; i < 16 * 128; i += 256) {
    int m = i >> 7, jj = i & 127;
    int b = bbase + m;
    if (b < 375) HLAST[(size_t)b * 128 + jj] = h1[i];
  }
}

// ---------------- head: logits + sum over k + log_softmax ----------------
__global__ void k_head(const float* __restrict__ HLAST, const float* __restrict__ tw,
                       const float* __restrict__ tb, float* __restrict__ out) {
  int q = blockIdx.x, l = threadIdx.x;
  __shared__ float lg[5];
  for (int ni = 0; ni < 5; ++ni) {
    float acc = 0.f;
    for (int i = l; i < 640; i += 32) {  // 5 shots x 128 dims
      int ki = i >> 7, d = i & 127;
      acc += HLAST[(size_t)(q * 25 + ni * 5 + ki) * 128 + d] * tw[d];
    }
    for (int o = 16; o > 0; o >>= 1) acc += __shfl_xor(acc, o, 32);
    if (l == 0) lg[ni] = acc + 5.f * tb[0];
  }
  __syncthreads();
  if (l == 0) {
    float m = -__builtin_inff();
    for (int i = 0; i < 5; ++i) m = fmaxf(m, lg[i]);
    float s = 0.f;
    for (int i = 0; i < 5; ++i) s += expf(lg[i] - m);
    float ls = logf(s);
    for (int i = 0; i < 5; ++i) out[q * 5 + i] = lg[i] - m - ls;
  }
}

extern "C" void kernel_launch(void* const* d_in, const int* in_sizes, int n_in,
                              void* d_out, int out_size, void* d_ws, size_t ws_size,
                              hipStream_t stream) {
  (void)in_sizes; (void)n_in; (void)out_size; (void)ws_size;
  const float* support = (const float*)d_in[0];
  const float* query   = (const float*)d_in[1];
  const float* w1 = (const float*)d_in[2];  const float* b1 = (const float*)d_in[3];
  const float* w2 = (const float*)d_in[4];  const float* b2 = (const float*)d_in[5];
  const float* w3 = (const float*)d_in[6];  const float* b3 = (const float*)d_in[7];
  const float* w4 = (const float*)d_in[8];  const float* b4 = (const float*)d_in[9];
  const float* g1 = (const float*)d_in[10]; const float* be1 = (const float*)d_in[11];
  const float* g2 = (const float*)d_in[12]; const float* be2 = (const float*)d_in[13];
  const float* g3 = (const float*)d_in[14]; const float* be3 = (const float*)d_in[15];
  const float* g4 = (const float*)d_in[16]; const float* be4 = (const float*)d_in[17];
  const float* att_w = (const float*)d_in[18]; const float* att_b = (const float*)d_in[19];
  const float* wi0 = (const float*)d_in[20]; const float* wh0 = (const float*)d_in[21];
  const float* bi0 = (const float*)d_in[22]; const float* bh0 = (const float*)d_in[23];
  const float* wi1 = (const float*)d_in[24]; const float* wh1 = (const float*)d_in[25];
  const float* bi1 = (const float*)d_in[26]; const float* bh1 = (const float*)d_in[27];
  const float* tw = (const float*)d_in[28]; const float* tb = (const float*)d_in[29];
  float* ws = (float*)d_ws;

  // workspace layout (float elements), with reuse:
  const size_t O_A1   = 0;          // 40*32*56*56 = 4,014,080 (later: C3, then WH/WQ/SC/HLAST)
  const size_t O_C2   = 4014080;    // 40*64*56*56 = 8,028,160 (later: C4, then GI0)
  const size_t O_A2   = 12042240;   // 40*64*28*28 = 2,007,040 (later: A4/pool4)
  const size_t O_A3   = 14049280;   // 40*128*14*14 = 1,003,520
  const size_t O_ATTD = 15052800;   // 15*25*49*256 = 4,704,000
  const size_t O_X    = 19756800;   // 375*49*512 = 9,408,000 -> peak 29,164,800 floats
  const size_t O_C3 = O_A1, O_C4 = O_C2, O_A4 = O_A2, O_GI0 = O_C2;
  const size_t O_WH = O_A1;                // 1225*64 = 78,400
  const size_t O_WQ = O_A1 + 78400;        // 735*64  = 47,040
  const size_t O_SC = O_A1 + 131072;       // 900,375
  const size_t O_HL = O_A1 + 1100000;      // 375*128 = 48,000

  // block 1
  { int total = 40 * 32 * 56 * 56;
    k_conv1_pool<<<dim3((total + 255) / 256), dim3(256), 0, stream>>>(
        support, query, w1, b1, g1, be1, ws + O_A1); }
  // block 2: 32->64 @56x56, pool -> 28
  k_conv_wmma<<<dim3(98, 4, 40), dim3(32), 0, stream>>>(
      ws + O_A1, w2, b2, g2, be2, ws + O_C2, 32, 64, 56, 56);
  { int total = 40 * 64 * 28 * 28;
    k_pool2<<<dim3((total + 255) / 256), dim3(256), 0, stream>>>(ws + O_C2, ws + O_A2, 64, 56, 56, 40); }
  // block 3: 64->128 @28x28, pool -> 14
  k_conv_wmma<<<dim3(25, 8, 40), dim3(32), 0, stream>>>(
      ws + O_A2, w3, b3, g3, be3, ws + O_C3, 64, 128, 28, 28);
  { int total = 40 * 128 * 14 * 14;
    k_pool2<<<dim3((total + 255) / 256), dim3(256), 0, stream>>>(ws + O_C3, ws + O_A3, 128, 28, 28, 40); }
  // block 4: 128->256 @14x14, pool -> 7
  k_conv_wmma<<<dim3(7, 16, 40), dim3(32), 0, stream>>>(
      ws + O_A3, w4, b4, g4, be4, ws + O_C4, 128, 256, 14, 14);
  { int total = 40 * 256 * 7 * 7;
    k_pool2<<<dim3((total + 255) / 256), dim3(256), 0, stream>>>(ws + O_C4, ws + O_A4, 256, 14, 14, 40); }
  // attention projections: Wh (support, 1225 rows), Wq (query, 735 rows)
  k_att_proj<<<dim3(4, 77), dim3(32), 0, stream>>>(ws + O_A4, att_w, att_b, ws + O_WH, 0, 1225);
  k_att_proj<<<dim3(4, 46), dim3(32), 0, stream>>>(ws + O_A4, att_w, att_b, ws + O_WQ, 25, 735);
  // scores + softmax
  { int total = 15 * 25 * 49 * 49;
    k_scores<<<dim3((total + 255) / 256), dim3(256), 0, stream>>>(ws + O_WQ, ws + O_WH, ws + O_SC); }
  k_softmax49<<<dim3(15 * 25 * 49), dim3(32), 0, stream>>>(ws + O_SC);
  // attended = att @ hf
  k_attended<<<dim3(16, 4, 375), dim3(32), 0, stream>>>(ws + O_SC, ws + O_A4, ws + O_ATTD);
  // GRU input tensor + layer-0 input projection (TDM-staged A panel)
  { size_t total = (size_t)375 * 49 * 512;
    k_build_x<<<dim3((unsigned)((total + 255) / 256)), dim3(256), 0, stream>>>(
        ws + O_ATTD, ws + O_A4, ws + O_X); }
  k_gi0<<<dim3(1149), dim3(256), 0, stream>>>(ws + O_X, wi0, bi0, ws + O_GI0);
  // fused 2-layer GRU recurrence
  k_gru<<<dim3(24), dim3(256), 0, stream>>>(ws + O_GI0, wh0, bh0, wi1, bi1, wh1, bh1, ws + O_HL);
  // head
  k_head<<<dim3(15), dim3(32), 0, stream>>>(ws + O_HL, tw, tb, (float*)d_out);
}
